// LSTM_10402410791794
// MI455X (gfx1250) — matrix operand order repaired and checked by reading, not compiled, not run on hardware
//
#include <hip/hip_runtime.h>
#include <stdint.h>

typedef __attribute__((ext_vector_type(16))) _Float16 v16h;
typedef __attribute__((ext_vector_type(8)))  _Float16 v8h;
typedef __attribute__((ext_vector_type(8)))  float    v8f;
typedef __attribute__((ext_vector_type(4)))  unsigned u32x4;
typedef __attribute__((ext_vector_type(8)))  int      i32x8;
typedef __attribute__((ext_vector_type(4)))  int      i32x4;

#define SEQ   512
#define BSZ   256
#define EMB   200                 // K
#define HID   10
#define G4    40                  // N = 4*HID
#define KPAD  224                 // 7 x 32, zero-padded K
#define NKB   7                   // KPAD/32
#define BFRAG_ELEMS (NKB*3*32*16) // 10752 halfs, fragment-ordered W_ih
#define BFRAG_DWORDS (BFRAG_ELEMS/2) // 5376
#define GATES_BYTES ((size_t)SEQ*BSZ*G4*4)
#define WSTRIDE 204               // per-thread W_hh LDS stride (floats, 16B-aligned)
#define LN_EPS 1e-5f

#ifndef __has_builtin
#define __has_builtin(x) 0
#endif
#if defined(__AMDGCN__) && __has_builtin(__builtin_amdgcn_tensor_load_to_lds)
#define HAVE_TDM 1
#else
#define HAVE_TDM 0
#endif
#if defined(__has_include)
#if __has_include(<hip/amd_detail/amd_gfx1250_TDM.h>)
#define TDM_6ARG 1
#else
#define TDM_6ARG 0
#endif
#else
#define TDM_6ARG 0
#endif

__device__ __forceinline__ float sigmoidf_(float x) {
    return 1.0f / (1.0f + __expf(-x));
}

// ---------------------------------------------------------------------------
// Prep A: convert embedding table f32[vocab,200] -> f16[vocab,224], zero-pad.
// One dword (2 halfs) per thread; 200 is even so halves never straddle K=200.
// ---------------------------------------------------------------------------
__global__ __launch_bounds__(256)
void emb_to_f16(const float* __restrict__ embW,
                _Float16*    __restrict__ emb16,
                int vocab)
{
    const int i = blockIdx.x * 256 + threadIdx.x;     // dword index
    const int ndw = vocab * (KPAD / 2);
    if (i >= ndw) return;
    const int r = i / (KPAD / 2);
    const int k = (i % (KPAD / 2)) * 2;
    const float x0 = (k     < EMB) ? embW[(long)r * EMB + k]     : 0.f;
    const float x1 = (k + 1 < EMB) ? embW[(long)r * EMB + k + 1] : 0.f;
    emb16[(long)r * KPAD + k]     = (_Float16)x0;
    emb16[(long)r * KPAD + k + 1] = (_Float16)x1;
}

// ---------------------------------------------------------------------------
// Prep B: pack W_ih (f32 [40,200]) into fragment-ordered f16, zero-padded to
// [7 kb][3 nt][32 lane][16 e], plus padded 48-entry bias table.
// ---------------------------------------------------------------------------
__global__ __launch_bounds__(256)
void lstm_prep(const float* __restrict__ Wih,
               const float* __restrict__ bih,
               const float* __restrict__ bhh,
               _Float16*    __restrict__ bfrag,
               float*       __restrict__ bias48)
{
    for (int i = threadIdx.x; i < BFRAG_ELEMS; i += 256) {
        const int e    = i & 15;
        const int lane = (i >> 4) & 31;
        const int nt   = (i >> 9) % 3;
        const int kbi  = i / 1536;
        const int n    = nt * 16 + (lane & 15);
        const int k    = kbi * 32 + ((lane >= 16) ? 16 : 0) + e;
        const float v  = (n < G4 && k < EMB) ? Wih[n * EMB + k] : 0.f;
        bfrag[i] = (_Float16)v;
    }
    if (threadIdx.x < 48) {
        const int n = threadIdx.x;
        bias48[n] = (n < G4) ? (bih[n] + bhh[n]) : 0.f;
    }
}

// ---------------------------------------------------------------------------
// Kernel 1: gates_x = gather(emb16, tok) @ W_ih^T + bias
// M=131072, K=224 (pre-padded), N=48 (padded); v_wmma_f32_16x16x32_f16.
// W_ih frags staged into LDS by the Tensor Data Mover; hot loop is pure
// data movement + WMMA: 2x global_load_b128 (A, f16, no guards/cvts),
// 2x3 ds_load_b128 (B), 3x v_wmma per K-block.
// ---------------------------------------------------------------------------
__global__ __launch_bounds__(256)
void lstm_gates_gemm(const int*      __restrict__ tokens,   // [BSZ, SEQ]
                     const _Float16* __restrict__ emb16,    // [vocab, KPAD]
                     const _Float16* __restrict__ bfrag,    // packed W_ih
                     const float*    __restrict__ bias48,
                     float*          __restrict__ gates)    // [SEQ, BSZ, G4]
{
    __shared__ __align__(32) _Float16 sB[BFRAG_ELEMS];   // 21504 B

#if HAVE_TDM
    if (threadIdx.x == 0) {
        const unsigned lds_off = (unsigned)(uintptr_t)&sB[0];
        const unsigned long long ga = (unsigned long long)(uintptr_t)bfrag;
        // D# group 0: count=1 | lds_addr | global_addr | type=2
        u32x4 g0 = { 1u, lds_off, (unsigned)ga,
                     (unsigned)((ga >> 32) & 0x01FFFFFFu) | 0x80000000u };
        // D# group 1: data_size=4B; tensor_dim0=5376, tensor_dim1=1;
        //             tile_dim0=5376, tile_dim1=1; dim0_stride=5376
        i32x8 g1 = { 0x00020000, 0x15000000, 0x00010000, 0x15000000,
                     1, BFRAG_DWORDS, 0, 0 };
        i32x4 gz = { 0, 0, 0, 0 };
#if TDM_6ARG
        i32x8 gz8 = { 0, 0, 0, 0, 0, 0, 0, 0 };
        __builtin_amdgcn_tensor_load_to_lds(g0, g1, gz, gz, gz8, 0);
#else
        __builtin_amdgcn_tensor_load_to_lds(g0, g1, gz, gz, 0);
#endif
#if __has_builtin(__builtin_amdgcn_s_wait_tensorcnt)
        __builtin_amdgcn_s_wait_tensorcnt(0);
#else
        asm volatile("s_wait_tensorcnt 0x0" ::: "memory");
#endif
    }
#else
    for (int i = threadIdx.x; i < BFRAG_DWORDS; i += 256)
        ((unsigned*)sB)[i] = ((const unsigned*)bfrag)[i];
#endif
    __syncthreads();

    const int lane  = threadIdx.x & 31;
    const int wave  = threadIdx.x >> 5;
    const int mtile = blockIdx.x * 8 + wave;       // 8192 M tiles total
    const int hi    = lane >> 4;
    const int nlane = lane & 15;

    const int mrow = mtile * 16 + nlane;           // m = s*BSZ + b
    const int s    = mrow >> 8;
    const int b    = mrow & 255;
    const int tok  = tokens[b * SEQ + s];
    const _Float16* arow = emb16 + (long)tok * KPAD;

    v8f acc0 = {}; v8f acc1 = {}; v8f acc2 = {};

    #pragma unroll
    for (int kbi = 0; kbi < NKB; ++kbi) {
        // A fragment (contiguous K order k = 16*hi + e): two aligned 16B loads
        const int kh = kbi * 32 + (hi ? 16 : 0);
        const v8h alo = *(const v8h*)(arow + kh);        // K kh..kh+7
        const v8h ahi = *(const v8h*)(arow + kh + 8);    // K kh+8..kh+15
        const v16h a = __builtin_shufflevector(alo, ahi,
                0, 1, 2, 3, 4, 5, 6, 7, 8, 9, 10, 11, 12, 13, 14, 15);
        // B fragments: single aligned 32B LDS read each
        const v16h bf0 = *(const v16h*)&sB[((kbi * 3 + 0) * 32 + lane) * 16];
        const v16h bf1 = *(const v16h*)&sB[((kbi * 3 + 1) * 32 + lane) * 16];
        const v16h bf2 = *(const v16h*)&sB[((kbi * 3 + 2) * 32 + lane) * 16];
        acc0 = __builtin_amdgcn_wmma_f32_16x16x32_f16(false, a, false, bf0,
                                                      (short)0, acc0, false, false);
        acc1 = __builtin_amdgcn_wmma_f32_16x16x32_f16(false, a, false, bf1,
                                                      (short)0, acc1, false, false);
        acc2 = __builtin_amdgcn_wmma_f32_16x16x32_f16(false, a, false, bf2,
                                                      (short)0, acc2, false, false);
    }

    // store D (lane = N, VGPR j = M row (+8 for hi half))
    const int mbase = mtile * 16 + (hi ? 8 : 0);
    #pragma unroll
    for (int nt = 0; nt < 3; ++nt) {
        const v8f acc = (nt == 0) ? acc0 : (nt == 1) ? acc1 : acc2;
        const int n = nt * 16 + nlane;
        const float bias = bias48[n];                 // padded: no guard
        if (n < G4) {
            #pragma unroll
            for (int j = 0; j < 8; ++j)
                gates[(long)(mbase + j) * G4 + n] = acc[j] + bias;
        }
    }
}

// ---------------------------------------------------------------------------
// Kernel 2: sequential LSTM scan (512 steps) + fused head.
// 8 blocks x 64 threads; block owns 32 batch rows; 2 threads per row, each
// owning 5 hidden units -> no cross-thread gate traffic. W_hh is replicated
// per-thread in LDS, in exact use order, so each step reads it with
// ds_load_b128 (50/step instead of 200 scalar reads). h double-buffered in
// LDS, one barrier per step; next gate slab prefetched.
// ---------------------------------------------------------------------------
__global__ __launch_bounds__(64)
void lstm_scan_head(const float* __restrict__ gates,  // [SEQ, BSZ, G4]
                    const float* __restrict__ h0,
                    const float* __restrict__ c0,
                    const float* __restrict__ Whh,    // [G4, HID]
                    const float* __restrict__ W1,
                    const float* __restrict__ b1,
                    const float* __restrict__ lng,
                    const float* __restrict__ lnb,
                    const float* __restrict__ W2,
                    const float* __restrict__ b2,
                    float*       __restrict__ out)    // [BSZ]
{
    __shared__ __align__(16) float sWT[64 * WSTRIDE]; // 52.2 KB (of 320 KB)
    __shared__ float sh[2][32 * HID];

    const int t  = threadIdx.x;
    const int bl = t >> 1;
    const int p  = t & 1;
    const int b  = blockIdx.x * 32 + bl;
    const int wbase = t * WSTRIDE;

    // Per-thread W_hh copy in use order: [q][class][jj], classes i,f,g,o.
    #pragma unroll
    for (int q = 0; q < 5; ++q)
        #pragma unroll
        for (int cls = 0; cls < 4; ++cls)
            for (int jj = 0; jj < HID; ++jj)
                sWT[wbase + q * 40 + cls * 10 + jj] =
                    Whh[(cls * 10 + p * 5 + q) * HID + jj];

    float c[5], hn[5];
    #pragma unroll
    for (int q = 0; q < 5; ++q) {
        const int j = p * 5 + q;
        sh[0][bl * HID + j] = h0[b * HID + j];
        c[q] = c0[b * HID + j];
    }
    __syncthreads();

    const float* gb = gates + (long)b * G4;
    int cur = 0;
    for (int s = 0; s < SEQ; ++s) {
        const float* g = gb + (long)s * (BSZ * G4);
        if (s + 1 < SEQ)
            __builtin_prefetch(g + BSZ * G4, 0, 0);   // global_prefetch_b8

        float hreg[HID];
        #pragma unroll
        for (int j = 0; j < HID; ++j) hreg[j] = sh[cur][bl * HID + j];

        #pragma unroll
        for (int q = 0; q < 5; ++q) {
            const int j = p * 5 + q;
            float xi = g[j];
            float xf = g[HID + j];
            float xg = g[2 * HID + j];
            float xo = g[3 * HID + j];
            // vector-load this q's 40 W values (10x ds_load_b128)
            float wv[40];
            #pragma unroll
            for (int cc = 0; cc < 10; ++cc)
                *(float4*)&wv[cc * 4] =
                    *(const float4*)&sWT[wbase + q * 40 + cc * 4];
            #pragma unroll
            for (int jj = 0; jj < HID; ++jj) {
                const float hv = hreg[jj];
                xi += hv * wv[jj];
                xf += hv * wv[10 + jj];
                xg += hv * wv[20 + jj];
                xo += hv * wv[30 + jj];
            }
            xi = sigmoidf_(xi);
            xf = sigmoidf_(xf);
            xg = tanhf(xg);
            xo = sigmoidf_(xo);
            c[q]  = xf * c[q] + xi * xg;
            hn[q] = xo * tanhf(c[q]);
        }
        const int nxt = cur ^ 1;
        #pragma unroll
        for (int q = 0; q < 5; ++q) sh[nxt][bl * HID + p * 5 + q] = hn[q];
        __syncthreads();
        cur = nxt;
    }

    if (p == 0) {
        float h[HID];
        #pragma unroll
        for (int j = 0; j < HID; ++j) h[j] = sh[cur][bl * HID + j];
        float z[5];
        float mu = 0.f;
        #pragma unroll
        for (int k = 0; k < 5; ++k) {
            float zz = b1[k];
            #pragma unroll
            for (int j = 0; j < HID; ++j) zz += h[j] * W1[k * HID + j];
            z[k] = zz; mu += zz;
        }
        mu *= 0.2f;
        float var = 0.f;
        #pragma unroll
        for (int k = 0; k < 5; ++k) { const float d = z[k] - mu; var += d * d; }
        var *= 0.2f;
        const float inv = rsqrtf(var + LN_EPS);
        float acc = b2[0];
        #pragma unroll
        for (int k = 0; k < 5; ++k)
            acc += ((z[k] - mu) * inv * lng[k] + lnb[k]) * W2[k];
        out[b] = sigmoidf_(acc);
    }
}

// ---------------------------------------------------------------------------
extern "C" void kernel_launch(void* const* d_in, const int* in_sizes, int n_in,
                              void* d_out, int out_size, void* d_ws, size_t ws_size,
                              hipStream_t stream) {
    const int*   tokens = (const int*)  d_in[0];
    const float* h0     = (const float*)d_in[1];
    const float* c0     = (const float*)d_in[2];
    const float* embW   = (const float*)d_in[3];
    const float* Wih    = (const float*)d_in[4];
    const float* Whh    = (const float*)d_in[5];
    const float* bih    = (const float*)d_in[6];
    const float* bhh    = (const float*)d_in[7];
    const float* W1     = (const float*)d_in[8];
    const float* b1     = (const float*)d_in[9];
    const float* lng    = (const float*)d_in[10];
    const float* lnb    = (const float*)d_in[11];
    const float* W2     = (const float*)d_in[12];
    const float* b2     = (const float*)d_in[13];

    const int vocab = in_sizes[3] / EMB;

    float*    gates  = (float*)d_ws;                              // 20.97 MB
    _Float16* emb16  = (_Float16*)((char*)d_ws + GATES_BYTES);    // vocab*224*2
    const size_t emb16_bytes = (size_t)vocab * KPAD * 2;
    _Float16* bfrag  = (_Float16*)((char*)d_ws + GATES_BYTES + emb16_bytes);
    float*    bias48 = (float*)((char*)d_ws + GATES_BYTES + emb16_bytes
                                + BFRAG_ELEMS * 2);
    float*    outp   = (float*)d_out;

    const int ndw = vocab * (KPAD / 2);
    emb_to_f16<<<(ndw + 255) / 256, 256, 0, stream>>>(embW, emb16, vocab);

    lstm_prep<<<1, 256, 0, stream>>>(Wih, bih, bhh, bfrag, bias48);

    lstm_gates_gemm<<<(SEQ * BSZ) / (16 * 8), 256, 0, stream>>>(
        tokens, emb16, bfrag, bias48, gates);

    lstm_scan_head<<<BSZ / 32, 64, 0, stream>>>(
        gates, h0, c0, Whh, W1, b1, lng, lnb, W2, b2, outp);
}
